// DTANet_46428596470259
// MI455X (gfx1250) — compile-verified, matrix operand order checked by reading, not run on hardware
//
#include <hip/hip_runtime.h>
#include <stdint.h>

// ---------------------------------------------------------------------------
// CDNA5 (gfx1250) WMMA types
// ---------------------------------------------------------------------------
typedef __attribute__((ext_vector_type(16))) __bf16        v16bf;
typedef __attribute__((ext_vector_type(8)))  float         v8f;
typedef __attribute__((ext_vector_type(4)))  unsigned int  u32x4;

union FragBF { v16bf v; u32x4 q[2]; };

__device__ __forceinline__ unsigned short f2bf(float f) {
  // round-to-nearest-even fp32 -> bf16
  unsigned x = __float_as_uint(f);
  unsigned r = (x + 0x7FFFu + ((x >> 16) & 1u)) >> 16;
  return (unsigned short)r;
}

// monotone float<->uint encoding for atomicMax-based segment_max
__device__ __forceinline__ unsigned encf(float f) {
  unsigned u = __float_as_uint(f);
  return (u & 0x80000000u) ? ~u : (u | 0x80000000u);
}
__device__ __forceinline__ float decf(unsigned u) {
  return __uint_as_float((u & 0x80000000u) ? (u & 0x7FFFFFFFu) : ~u);
}
#define ENC_NEG_INF 0x007FFFFFu   // encf(-inf)

#define TBM 128
#define TBN 128
#define TBK 32
#define LDT 40   // padded LDS row stride in ushort (80B: multiple of 16B)
#define TILE_US (TBM * LDT)  // ushorts per tile buffer

// ---------------------------------------------------------------------------
// Aligned WMMA GEMM fast path: requires N%128==0, M%128==0, K%32==0.
// C[N x M] = A[N x K] @ B[K x M] + bias[M], fp32 global, bf16 WMMA inside.
// Block = 256 threads (8 waves), 128x128 tile, K-step 32.
// Double-buffered LDS pipeline: global loads of tile t+1 issue before the
// WMMAs of tile t, so s_wait_loadcnt lands after the matrix work.
// ---------------------------------------------------------------------------
__global__ __launch_bounds__(256)
void wmma_gemm_aligned(const float* __restrict__ A, const float* __restrict__ B,
                       const float* __restrict__ bias, float* __restrict__ C,
                       int N, int K, int M) {
  __shared__ __align__(16) unsigned short As[2 * TILE_US];
  __shared__ __align__(16) unsigned short Bs[2 * TILE_US];

  const int tid  = threadIdx.x;
  const int wave = tid >> 5;
  const int lane = tid & 31;
  const int wr   = wave & 3;   // row group of 32
  const int wc   = wave >> 2;  // col group of 64
  const int h    = lane >> 4;  // lane half
  const int ln   = lane & 15;

  const int rowbase = blockIdx.y * TBM;
  const int colbase = blockIdx.x * TBN;

  // staging maps (branch-free, fully coalesced 16B loads)
  const int ar0 = tid >> 3;        // A: row within tile   (8 float4 per row)
  const int ac4 = (tid & 7) * 4;   //    col (float) within 32-wide K slab
  const int bk0 = tid >> 5;        // B: k within slab     (32 float4 per row)
  const int bn4 = (tid & 31) * 4;  //    col (float) within 128-wide N slab

  v8f zero = {0.f, 0.f, 0.f, 0.f, 0.f, 0.f, 0.f, 0.f};
  v8f acc[2][4];
#pragma unroll
  for (int i = 0; i < 2; ++i)
#pragma unroll
    for (int j = 0; j < 4; ++j) acc[i][j] = zero;

  const int nsteps = K / TBK;
  float4 ra[4], rb[4];

  // ---- prologue: load + stage tile 0 into buffer 0 ----
#pragma unroll
  for (int j = 0; j < 4; ++j)
    ra[j] = *(const float4*)&A[(size_t)(rowbase + ar0 + j * 32) * K + ac4];
#pragma unroll
  for (int j = 0; j < 4; ++j)
    rb[j] = *(const float4*)&B[(size_t)(bk0 + j * 8) * M + colbase + bn4];
#pragma unroll
  for (int j = 0; j < 4; ++j) {
    ushort4 b;
    b.x = f2bf(ra[j].x); b.y = f2bf(ra[j].y);
    b.z = f2bf(ra[j].z); b.w = f2bf(ra[j].w);
    *(ushort4*)&As[(ar0 + j * 32) * LDT + ac4] = b;
    int kk = bk0 + j * 8;
    Bs[(bn4 + 0) * LDT + kk] = f2bf(rb[j].x);
    Bs[(bn4 + 1) * LDT + kk] = f2bf(rb[j].y);
    Bs[(bn4 + 2) * LDT + kk] = f2bf(rb[j].z);
    Bs[(bn4 + 3) * LDT + kk] = f2bf(rb[j].w);
  }

  for (int t = 0; t < nsteps; ++t) {
    __syncthreads();
    const int cur = t & 1;
    const int nxt = cur ^ 1;
    const int k1 = (t + 1) * TBK;
    const bool more = (t + 1) < nsteps;

    // ---- issue next tile's global loads (no wait yet) ----
    if (more) {
#pragma unroll
      for (int j = 0; j < 4; ++j)
        ra[j] = *(const float4*)&A[(size_t)(rowbase + ar0 + j * 32) * K + k1 + ac4];
#pragma unroll
      for (int j = 0; j < 4; ++j)
        rb[j] = *(const float4*)&B[(size_t)(k1 + bk0 + j * 8) * M + colbase + bn4];
      // warm L2 for tile t+2
      if (k1 + TBK < K) {
        __builtin_prefetch(&A[(size_t)(rowbase + (tid >> 1)) * K + k1 + TBK], 0, 0);
        __builtin_prefetch(&B[(size_t)(k1 + TBK + (tid >> 7)) * M + colbase + (tid & 127)], 0, 0);
      }
    }

    // ---- compute on current buffer (hides global-load latency) ----
    const unsigned short* Ac = &As[cur * TILE_US];
    const unsigned short* Bc = &Bs[cur * TILE_US];
    FragBF af[2], bf[4];
#pragma unroll
    for (int rf = 0; rf < 2; ++rf) {
      int ar = wr * 32 + rf * 16 + ln;
      af[rf].q[0] = *(const u32x4*)&Ac[ar * LDT + 8 * h];
      af[rf].q[1] = *(const u32x4*)&Ac[ar * LDT + 16 + 8 * h];
    }
#pragma unroll
    for (int cf = 0; cf < 4; ++cf) {
      int bn = wc * 64 + cf * 16 + ln;
      bf[cf].q[0] = *(const u32x4*)&Bc[bn * LDT + 8 * h];
      bf[cf].q[1] = *(const u32x4*)&Bc[bn * LDT + 16 + 8 * h];
    }
#pragma unroll
    for (int rf = 0; rf < 2; ++rf)
#pragma unroll
      for (int cf = 0; cf < 4; ++cf)
        acc[rf][cf] = __builtin_amdgcn_wmma_f32_16x16x32_bf16(
            false, af[rf].v, false, bf[cf].v, (short)0, acc[rf][cf],
            false, false);

    // ---- convert + store next tile into other buffer ----
    if (more) {
      unsigned short* An = &As[nxt * TILE_US];
      unsigned short* Bn = &Bs[nxt * TILE_US];
#pragma unroll
      for (int j = 0; j < 4; ++j) {
        ushort4 b;
        b.x = f2bf(ra[j].x); b.y = f2bf(ra[j].y);
        b.z = f2bf(ra[j].z); b.w = f2bf(ra[j].w);
        *(ushort4*)&An[(ar0 + j * 32) * LDT + ac4] = b;
        int kk = bk0 + j * 8;
        Bn[(bn4 + 0) * LDT + kk] = f2bf(rb[j].x);
        Bn[(bn4 + 1) * LDT + kk] = f2bf(rb[j].y);
        Bn[(bn4 + 2) * LDT + kk] = f2bf(rb[j].z);
        Bn[(bn4 + 3) * LDT + kk] = f2bf(rb[j].w);
      }
    }
  }

  // ---- writeback, no guards (aligned) ----
#pragma unroll
  for (int rf = 0; rf < 2; ++rf) {
#pragma unroll
    for (int cf = 0; cf < 4; ++cf) {
      int gc = colbase + wc * 64 + cf * 16 + ln;
      float bb = bias[gc];
      int grb = rowbase + wr * 32 + rf * 16 + h * 8;
#pragma unroll
      for (int v = 0; v < 8; ++v)
        C[(size_t)(grb + v) * M + gc] = acc[rf][cf][v] + bb;
    }
  }
}

// ---------------------------------------------------------------------------
// Generic guarded WMMA GEMM (head MLP only: tiny N=32, unaligned M)
// ---------------------------------------------------------------------------
__global__ __launch_bounds__(256)
void wmma_gemm_generic(const float* __restrict__ A, const float* __restrict__ B,
                       const float* __restrict__ bias, float* __restrict__ C,
                       int N, int K, int M) {
  __shared__ __align__(16) unsigned short As[TILE_US];
  __shared__ __align__(16) unsigned short Bs[TILE_US];

  const int tid  = threadIdx.x;
  const int wave = tid >> 5;
  const int lane = tid & 31;
  const int wr   = wave & 3;
  const int wc   = wave >> 2;
  const int h    = lane >> 4;
  const int ln   = lane & 15;

  const int rowbase = blockIdx.y * TBM;
  const int colbase = blockIdx.x * TBN;

  v8f zero = {0.f, 0.f, 0.f, 0.f, 0.f, 0.f, 0.f, 0.f};
  v8f acc[2][4];
#pragma unroll
  for (int i = 0; i < 2; ++i)
#pragma unroll
    for (int j = 0; j < 4; ++j) acc[i][j] = zero;

  for (int k0 = 0; k0 < K; k0 += TBK) {
    for (int i = tid; i < TBM * TBK; i += 256) {
      int r = i >> 5, kk = i & 31;
      int gr = rowbase + r, gk = k0 + kk;
      float v = (gr < N && gk < K) ? A[(size_t)gr * K + gk] : 0.f;
      As[r * LDT + kk] = f2bf(v);
    }
    for (int i = tid; i < TBN * TBK; i += 256) {
      int kk = i >> 7, n = i & 127;
      int gk = k0 + kk, gc = colbase + n;
      float v = (gk < K && gc < M) ? B[(size_t)gk * M + gc] : 0.f;
      Bs[n * LDT + kk] = f2bf(v);
    }
    __syncthreads();

    FragBF af[2], bf[4];
#pragma unroll
    for (int rf = 0; rf < 2; ++rf) {
      int ar = wr * 32 + rf * 16 + ln;
      af[rf].q[0] = *(const u32x4*)&As[ar * LDT + 8 * h];
      af[rf].q[1] = *(const u32x4*)&As[ar * LDT + 16 + 8 * h];
    }
#pragma unroll
    for (int cf = 0; cf < 4; ++cf) {
      int bn = wc * 64 + cf * 16 + ln;
      bf[cf].q[0] = *(const u32x4*)&Bs[bn * LDT + 8 * h];
      bf[cf].q[1] = *(const u32x4*)&Bs[bn * LDT + 16 + 8 * h];
    }
#pragma unroll
    for (int rf = 0; rf < 2; ++rf)
#pragma unroll
      for (int cf = 0; cf < 4; ++cf)
        acc[rf][cf] = __builtin_amdgcn_wmma_f32_16x16x32_bf16(
            false, af[rf].v, false, bf[cf].v, (short)0, acc[rf][cf],
            false, false);
    __syncthreads();
  }

#pragma unroll
  for (int rf = 0; rf < 2; ++rf) {
#pragma unroll
    for (int cf = 0; cf < 4; ++cf) {
      int gc = colbase + wc * 64 + cf * 16 + ln;
#pragma unroll
      for (int v = 0; v < 8; ++v) {
        int gr = rowbase + wr * 32 + rf * 16 + h * 8 + v;
        if (gr < N && gc < M)
          C[(size_t)gr * M + gc] = acc[rf][cf][v] + bias[gc];
      }
    }
  }
}

// ---------------------------------------------------------------------------
// GATv2 graph kernels
// ---------------------------------------------------------------------------
__global__ void init_stats(unsigned* __restrict__ amax, float* __restrict__ den,
                           int n) {
  int i = blockIdx.x * blockDim.x + threadIdx.x;
  if (i < n) { amax[i] = ENC_NEG_INF; den[i] = 0.f; }
}

__global__ void bcast_bias(const float* __restrict__ bias,
                           float* __restrict__ out, int N, int F) {
  int i = blockIdx.x * blockDim.x + threadIdx.x;
  if (i < N * F) out[i] = bias[i % F];
}

// e_ij = att . leaky_relu(xl[src] + xr[dst] + eattr@We); atomicMax per dst
__global__ __launch_bounds__(256)
void gat_scores(const float* __restrict__ xl, const float* __restrict__ xr,
                const float* __restrict__ eattr, const float* __restrict__ We,
                const float* __restrict__ att,
                const int* __restrict__ src, const int* __restrict__ dst,
                float* __restrict__ scores, unsigned* __restrict__ amax,
                int E, int H, int D, int ED) {
  int e = blockIdx.x;
  if (e >= E) return;
  const int HD = H * D;
  const int s = src[e], d = dst[e];
  __shared__ float sh[16];
  if (threadIdx.x < H) sh[threadIdx.x] = 0.f;
  __syncthreads();
  float ea[8];
  for (int j = 0; j < ED; ++j) ea[j] = eattr[(size_t)e * ED + j];
  for (int f = threadIdx.x; f < HD; f += blockDim.x) {
    float ew = 0.f;
    for (int j = 0; j < ED; ++j) ew += ea[j] * We[(size_t)j * HD + f];
    float m = xl[(size_t)s * HD + f] + xr[(size_t)d * HD + f] + ew;
    m = (m > 0.f) ? m : 0.2f * m;          // leaky_relu, slope 0.2
    atomicAdd(&sh[f / D], m * att[f]);     // ds_add_f32 per-head dot
  }
  __syncthreads();
  if (threadIdx.x < H) {
    float v = sh[threadIdx.x];
    scores[(size_t)e * H + threadIdx.x] = v;
    atomicMax(&amax[(size_t)d * H + threadIdx.x], encf(v));
  }
}

__global__ void gat_expsum(const float* __restrict__ scores,
                           const unsigned* __restrict__ amax,
                           const int* __restrict__ dst,
                           float* __restrict__ aexp, float* __restrict__ den,
                           int E, int H) {
  int i = blockIdx.x * blockDim.x + threadIdx.x;
  if (i >= E * H) return;
  int e = i / H, hh = i - e * H;
  int d = dst[e];
  float a = expf(scores[i] - decf(amax[(size_t)d * H + hh]));
  aexp[i] = a;
  atomicAdd(&den[(size_t)d * H + hh], a);
}

// out[dst] += alpha * xl[src]
__global__ __launch_bounds__(256)
void gat_scatter(const float* __restrict__ xl, const float* __restrict__ aexp,
                 const float* __restrict__ den,
                 const int* __restrict__ src, const int* __restrict__ dst,
                 float* __restrict__ out, int E, int H, int D) {
  int e = blockIdx.x;
  if (e >= E) return;
  const int HD = H * D;
  const int s = src[e], d = dst[e];
  __shared__ float sa[16];
  if (threadIdx.x < H) {
    float dn = den[(size_t)d * H + threadIdx.x];
    sa[threadIdx.x] = aexp[(size_t)e * H + threadIdx.x] / dn;
  }
  __syncthreads();
  for (int f = threadIdx.x; f < HD; f += blockDim.x)
    atomicAdd(&out[(size_t)d * HD + f], xl[(size_t)s * HD + f] * sa[f / D]);
}

// ---------------------------------------------------------------------------
// Pool / head kernels
// ---------------------------------------------------------------------------
__global__ void mean_pool_k(const float* __restrict__ x, float* __restrict__ hcat,
                            int npg, int F, int outStride, int colOff) {
  int f = blockIdx.x * blockDim.x + threadIdx.x;
  int b = blockIdx.y;
  if (f >= F) return;
  float sum = 0.f;
  for (int n = 0; n < npg; ++n) sum += x[(size_t)(b * npg + n) * F + f];
  hcat[(size_t)b * outStride + colOff + f] = sum / (float)npg;
}

__global__ void bn_relu_k(float* __restrict__ x, const float* __restrict__ g,
                          const float* __restrict__ be, int Bn, int F) {
  int f = blockIdx.x * blockDim.x + threadIdx.x;
  if (f >= F) return;
  float mu = 0.f, ss = 0.f;
  for (int r = 0; r < Bn; ++r) {
    float v = x[(size_t)r * F + f];
    mu += v; ss += v * v;
  }
  mu /= (float)Bn;
  float var = ss / (float)Bn - mu * mu;
  float rs = rsqrtf(var + 1e-5f);
  float gg = g[f], bb = be[f];
  for (int r = 0; r < Bn; ++r) {
    float v = (x[(size_t)r * F + f] - mu) * rs * gg + bb;
    x[(size_t)r * F + f] = v > 0.f ? v : 0.f;
  }
}

__global__ void final_linear(const float* __restrict__ h,
                             const float* __restrict__ W,
                             const float* __restrict__ b,
                             float* __restrict__ out, int F) {
  int bi = blockIdx.x;
  __shared__ float red[256];
  float s = 0.f;
  for (int f = threadIdx.x; f < F; f += blockDim.x)
    s += h[(size_t)bi * F + f] * W[f];
  red[threadIdx.x] = s;
  __syncthreads();
  for (int st = 128; st > 0; st >>= 1) {
    if (threadIdx.x < st) red[threadIdx.x] += red[threadIdx.x + st];
    __syncthreads();
  }
  if (threadIdx.x == 0) out[bi] = red[0] + b[0];
}

// ---------------------------------------------------------------------------
// Host orchestration
// ---------------------------------------------------------------------------
static inline size_t al256(size_t x) { return (x + 255) & ~(size_t)255; }
static inline int cdiv(int a, int b) { return (a + b - 1) / b; }

extern "C" void kernel_launch(void* const* d_in, const int* in_sizes, int n_in,
                              void* d_out, int out_size, void* d_ws, size_t ws_size,
                              hipStream_t stream) {
  (void)in_sizes; (void)n_in; (void)out_size; (void)ws_size;
  // --- inputs (setup_inputs dict order, params recursed in insertion order) ---
  const float* drug_x  = (const float*)d_in[0];
  const int*   drug_ei = (const int*)  d_in[1];
  const float* drug_ea = (const float*)d_in[2];
  const float* tgt_x   = (const float*)d_in[4];
  const int*   tgt_ei  = (const int*)  d_in[5];
  const float* tgt_ea  = (const float*)d_in[6];
  const float *dWl = (const float*)d_in[8],  *dbl = (const float*)d_in[9];
  const float *dWr = (const float*)d_in[10], *dbr = (const float*)d_in[11];
  const float *dWe = (const float*)d_in[12], *datt = (const float*)d_in[13];
  const float *dbias = (const float*)d_in[14];
  const float *tWl = (const float*)d_in[15], *tbl = (const float*)d_in[16];
  const float *tWr = (const float*)d_in[17], *tbr = (const float*)d_in[18];
  const float *tWe = (const float*)d_in[19], *tatt = (const float*)d_in[20];
  const float *tbias = (const float*)d_in[21];
  const float *W1 = (const float*)d_in[22], *b1 = (const float*)d_in[23];
  const float *g1 = (const float*)d_in[24], *be1 = (const float*)d_in[25];
  const float *W2 = (const float*)d_in[26], *b2 = (const float*)d_in[27];
  const float *g2 = (const float*)d_in[28], *be2 = (const float*)d_in[29];
  const float *W3 = (const float*)d_in[30], *b3 = (const float*)d_in[31];
  const float *g3 = (const float*)d_in[32], *be3 = (const float*)d_in[33];
  const float *W4 = (const float*)d_in[34], *b4 = (const float*)d_in[35];

  const int H = 8, Bg = 32;
  const int ND = 1024, ED_ = 8192, HDd = 512,  Dd = 64,  edd = 7, npgD = 32;
  const int NT = 6400, ET  = 64000, HDt = 2560, Dt = 320, edt = 1, npgT = 200;

  // --- workspace layout ---
  char* ws = (char*)d_ws;
  size_t off = 0;
  auto take = [&](size_t bytes) { char* p = ws + off; off += al256(bytes); return p; };
  float*    tA      = (float*)   take((size_t)NT * HDt * 4);
  float*    tB      = (float*)   take((size_t)NT * HDt * 4);
  float*    tC      = (float*)   take((size_t)NT * HDt * 4);
  float*    tScores = (float*)   take((size_t)ET * H * 4);
  float*    tAexp   = (float*)   take((size_t)ET * H * 4);
  unsigned* tAmax   = (unsigned*)take((size_t)NT * H * 4);
  float*    tDen    = (float*)   take((size_t)NT * H * 4);
  float*    dA      = (float*)   take((size_t)ND * HDd * 4);
  float*    dB      = (float*)   take((size_t)ND * HDd * 4);
  float*    dC      = (float*)   take((size_t)ND * HDd * 4);
  float*    dScores = (float*)   take((size_t)ED_ * H * 4);
  float*    dAexp   = (float*)   take((size_t)ED_ * H * 4);
  unsigned* dAmax   = (unsigned*)take((size_t)ND * H * 4);
  float*    dDen    = (float*)   take((size_t)ND * H * 4);
  float*    hcat    = (float*)   take((size_t)Bg * 3072 * 4);
  float*    h1      = (float*)   take((size_t)Bg * 1280 * 4);
  float*    h2      = (float*)   take((size_t)Bg * 640 * 4);
  float*    h3      = (float*)   take((size_t)Bg * 320 * 4);

  auto run_branch = [&](int N, int E, int HD, int D, int ED, int,
                        const float* x0, const int* ei, const float* ea,
                        const float* Wl, const float* bl, const float* Wr,
                        const float* br, const float* We, const float* att,
                        const float* bias, float* bufA, float* bufB, float* bufC,
                        float* scores, float* aexp, unsigned* amax, float* den) {
    const int* src = ei;
    const int* dst = ei + E;
    dim3 ggrid(HD / TBN, N / TBM);   // branch dims are tile-aligned
    for (int l = 0; l < 3; ++l) {
      const float* x = (l == 0) ? x0 : bufA;                 // in_dim == HD
      const float* Wl_l = Wl + (size_t)l * HD * HD;
      const float* Wr_l = Wr + (size_t)l * HD * HD;
      const float* bl_l = bl + (size_t)l * HD;
      const float* br_l = br + (size_t)l * HD;
      const float* We_l = We + (size_t)l * ED * HD;
      const float* at_l = att + (size_t)l * HD;
      const float* bi_l = bias + (size_t)l * HD;
      wmma_gemm_aligned<<<ggrid, 256, 0, stream>>>(x, Wl_l, bl_l, bufB, N, HD, HD);
      wmma_gemm_aligned<<<ggrid, 256, 0, stream>>>(x, Wr_l, br_l, bufC, N, HD, HD);
      init_stats<<<cdiv(N * H, 256), 256, 0, stream>>>(amax, den, N * H);
      gat_scores<<<E, 256, 0, stream>>>(bufB, bufC, ea, We_l, at_l, src, dst,
                                        scores, amax, E, H, D, ED);
      bcast_bias<<<cdiv(N * HD, 256), 256, 0, stream>>>(bi_l, bufA, N, HD);
      gat_expsum<<<cdiv(E * H, 256), 256, 0, stream>>>(scores, amax, dst, aexp,
                                                       den, E, H);
      gat_scatter<<<E, 256, 0, stream>>>(bufB, aexp, den, src, dst, bufA, E, H, D);
    }
  };

  run_branch(ND, ED_, HDd, Dd, edd, npgD, drug_x, drug_ei, drug_ea,
             dWl, dbl, dWr, dbr, dWe, datt, dbias,
             dA, dB, dC, dScores, dAexp, dAmax, dDen);
  run_branch(NT, ET, HDt, Dt, edt, npgT, tgt_x, tgt_ei, tgt_ea,
             tWl, tbl, tWr, tbr, tWe, tatt, tbias,
             tA, tB, tC, tScores, tAexp, tAmax, tDen);

  // mean pool -> concat [B, 3072] (drug 0..511, target 512..3071)
  mean_pool_k<<<dim3(cdiv(HDd, 256), Bg), 256, 0, stream>>>(dA, hcat, npgD, HDd, 3072, 0);
  mean_pool_k<<<dim3(cdiv(HDt, 256), Bg), 256, 0, stream>>>(tA, hcat, npgT, HDt, 3072, HDd);

  // head MLP (tiny, guarded generic path)
  wmma_gemm_generic<<<dim3(cdiv(1280, TBN), 1), 256, 0, stream>>>(hcat, W1, b1, h1, Bg, 3072, 1280);
  bn_relu_k<<<cdiv(1280, 256), 256, 0, stream>>>(h1, g1, be1, Bg, 1280);
  wmma_gemm_generic<<<dim3(cdiv(640, TBN), 1), 256, 0, stream>>>(h1, W2, b2, h2, Bg, 1280, 640);
  bn_relu_k<<<cdiv(640, 256), 256, 0, stream>>>(h2, g2, be2, Bg, 640);
  wmma_gemm_generic<<<dim3(cdiv(320, TBN), 1), 256, 0, stream>>>(h2, W3, b3, h3, Bg, 640, 320);
  bn_relu_k<<<cdiv(320, 256), 256, 0, stream>>>(h3, g3, be3, Bg, 320);
  final_linear<<<Bg, 256, 0, stream>>>(h3, W4, b4, (float*)d_out, 320);
}
